// GNNs_MoE_89824946028716
// MI455X (gfx1250) — matrix operand order repaired and checked by reading, not compiled, not run on hardware
//
#include <hip/hip_runtime.h>
#include <stdint.h>

#define N_NODES   100000
#define N_EDGES   1600000
#define NUM_G     512
#define D_GRAPH   128
#define HID       256
#define N_EXP     5
#define INNER     1024
#define L_MLP     4
#define OUT_C     10

typedef __attribute__((ext_vector_type(16))) __bf16 v16bf;
typedef __attribute__((ext_vector_type(8)))  __bf16 v8bf;
typedef __attribute__((ext_vector_type(8)))  float  v8f;

// ---------------- CDNA5 async global->LDS copy ------------------------------
__device__ __forceinline__ void async_ld_b128(unsigned lds_off, const void* g) {
  asm volatile("global_load_async_to_lds_b128 %0, %1, off"
               :: "v"(lds_off), "v"((unsigned long long)(uintptr_t)g)
               : "memory");
}
__device__ __forceinline__ void wait_async0() {
  asm volatile("s_wait_asynccnt 0" ::: "memory");
}

// ---------------- small elementwise kernels ---------------------------------
__global__ void k_deg(const int* __restrict__ dst, float* __restrict__ deg, int E) {
  int i = blockIdx.x * blockDim.x + threadIdx.x;
  if (i < E) atomicAdd(&deg[dst[i]], 1.0f);
}

__global__ void k_deginv(const float* __restrict__ deg, float* __restrict__ dinv, int n) {
  int i = blockIdx.x * blockDim.x + threadIdx.x;
  if (i < n) { float d = deg[i]; dinv[i] = (d > 0.0f) ? (1.0f / d) : 0.0f; }
}

__global__ void k_embed(const int* __restrict__ feat, const float* __restrict__ emb,
                        float* __restrict__ h) {
  int wave = (blockIdx.x * blockDim.x + threadIdx.x) >> 5;
  int lane = threadIdx.x & 31;
  if (wave < N_NODES) {
    int f = feat[wave];
    float4 v = ((const float4*)(emb + (size_t)f * D_GRAPH))[lane];
    ((float4*)(h + (size_t)wave * D_GRAPH))[lane] = v;
  }
}

// transpose-convert: in is KxN row-major fp32, out is NxK row-major bf16
__global__ void k_convert_t(const float* __restrict__ in, __bf16* __restrict__ out,
                            int K, int N) {
  int i = blockIdx.x * blockDim.x + threadIdx.x;
  if (i < K * N) {
    int k = i / N, n = i % N;
    out[(size_t)n * K + k] = (__bf16)in[i];
  }
}

// pred_W: 256x10 fp32 -> 16x256 bf16 (transposed + padded to 16 cols)
__global__ void k_convert_pred(const float* __restrict__ in, __bf16* __restrict__ out) {
  int i = blockIdx.x * blockDim.x + threadIdx.x;           // 16*256
  if (i < 16 * HID) {
    int n = i / HID, k = i % HID;
    out[i] = (__bf16)((n < OUT_C) ? in[k * OUT_C + n] : 0.0f);
  }
}

__global__ void k_edge_scatter(const float* __restrict__ h, const int* __restrict__ src,
                               const int* __restrict__ dst, float* __restrict__ agg, int E) {
  int wave  = (blockIdx.x * blockDim.x + threadIdx.x) >> 5;
  int lane  = threadIdx.x & 31;
  int nwave = (gridDim.x * blockDim.x) >> 5;
  for (int e = wave; e < E; e += nwave) {
    int e2 = e + nwave;
    if (e2 < E) __builtin_prefetch(h + (size_t)src[e2] * D_GRAPH + lane * 4, 0, 0);
    int s = src[e], d = dst[e];
    float4 v = ((const float4*)(h + (size_t)s * D_GRAPH))[lane];
    float* b = agg + (size_t)d * D_GRAPH + lane * 4;
    atomicAdd(b + 0, v.x); atomicAdd(b + 1, v.y);
    atomicAdd(b + 2, v.z); atomicAdd(b + 3, v.w);
  }
}

__global__ void k_scale_agg(float* __restrict__ agg, const float* __restrict__ dinv) {
  int i = blockIdx.x * blockDim.x + threadIdx.x;           // N_NODES * D
  if (i < N_NODES * D_GRAPH) agg[i] *= dinv[i >> 7];
}

__global__ void k_pool_accum(const float* __restrict__ h, const int* __restrict__ gid,
                             float* __restrict__ hgsum, float* __restrict__ cnt) {
  int wave = (blockIdx.x * blockDim.x + threadIdx.x) >> 5;
  int lane = threadIdx.x & 31;
  if (wave < N_NODES) {
    int g = gid[wave];
    float4 v = ((const float4*)(h + (size_t)wave * D_GRAPH))[lane];
    float* b = hgsum + (size_t)g * D_GRAPH + lane * 4;
    atomicAdd(b + 0, v.x); atomicAdd(b + 1, v.y);
    atomicAdd(b + 2, v.z); atomicAdd(b + 3, v.w);
    if (lane == 0) atomicAdd(&cnt[g], 1.0f);
  }
}

__global__ void k_pool_final(const float* __restrict__ hgsum, const float* __restrict__ cnt,
                             float* __restrict__ hg) {
  int i = blockIdx.x * blockDim.x + threadIdx.x;           // NUM_G * D
  if (i < NUM_G * D_GRAPH) hg[i] = hgsum[i] / fmaxf(cnt[i >> 7], 1.0f);
}

// ---------------- WMMA fragment helpers -------------------------------------
// A fragment (16x32 bf16) from a bf16 row-major [16 x 128] LDS tile.
// Per ISA 7.12.2: lane m = l&15; k-half = (l>>4)*8 within each 16-k group.
// That is exactly two contiguous 8-element (16 B) chunks per lane.
__device__ __forceinline__ v16bf frag_a_lds(const __bf16* As, int mLocal, int k0, int kha) {
  const __bf16* row = As + mLocal * D_GRAPH + k0;
  v8bf lo = *(const v8bf*)(row + kha);
  v8bf hi = *(const v8bf*)(row + 16 + kha);
  v16bf a;
#pragma unroll
  for (int r = 0; r < 8; ++r) { a[r] = lo[r]; a[8 + r] = hi[r]; }
  return a;
}

// A fragment built from fp32 row-major A with optional leaky-relu pre-act.
__device__ __forceinline__ v16bf frag_a_f32(const float* __restrict__ A, size_t lda,
                                            int mA, int k0, int kha, int preact) {
  const float* row = A + (size_t)mA * lda + k0;
  v16bf a;
#pragma unroll
  for (int r = 0; r < 4; ++r) {
    float x0 = row[kha + 2 * r];
    float x1 = row[kha + 2 * r + 1];
    float x2 = row[16 + kha + 2 * r];
    float x3 = row[16 + kha + 2 * r + 1];
    if (preact == 1) {
      x0 = (x0 < 0.f) ? 0.01f * x0 : x0;
      x1 = (x1 < 0.f) ? 0.01f * x1 : x1;
      x2 = (x2 < 0.f) ? 0.01f * x2 : x2;
      x3 = (x3 < 0.f) ? 0.01f * x3 : x3;
    }
    a[2 * r]     = (__bf16)x0; a[2 * r + 1] = (__bf16)x1;
    a[2 * r + 8] = (__bf16)x2; a[2 * r + 9] = (__bf16)x3;
  }
  return a;
}

// B fragment (32x16 bf16) from transposed (NxK row-major) bf16 weights:
// lane n = l&15; k-half = (l>>4)*8 within each 16-k group -> two 16 B chunks.
__device__ __forceinline__ v16bf frag_b_vec(const __bf16* __restrict__ Bt, size_t ldk,
                                            int nB, int k0, int khb) {
  const __bf16* row = Bt + (size_t)nB * ldk + k0 + khb;
  v8bf lo = *(const v8bf*)(row);
  v8bf hi = *(const v8bf*)(row + 16);
  v16bf b;
#pragma unroll
  for (int r = 0; r < 8; ++r) { b[r] = lo[r]; b[8 + r] = hi[r]; }
  return b;
}

// ---------------- GraphSAGE conv: relu(h@Wself + agg@Wneigh + b) ------------
__global__ void __launch_bounds__(256)
k_conv(const float* __restrict__ hin, const float* __restrict__ agg,
       const __bf16* __restrict__ wselfT, const __bf16* __restrict__ wneighT,
       const float* __restrict__ bias, float* __restrict__ hout) {
  __shared__ __align__(16) __bf16 WsT[D_GRAPH * D_GRAPH];   // 32 KB
  __shared__ __align__(16) __bf16 WnT[D_GRAPH * D_GRAPH];   // 32 KB
  __shared__ __align__(16) __bf16 Asf[16 * D_GRAPH];        // 4 KB: h tile (bf16)
  __shared__ __align__(16) __bf16 Anb[16 * D_GRAPH];        // 4 KB: agg tile (bf16)

  int m0 = blockIdx.x * 16;
  unsigned t = threadIdx.x;

  // async-stage both (pre-transposed bf16) weight matrices into LDS
  unsigned ldsS = (unsigned)(uintptr_t)(&WsT[0]);
  unsigned ldsN = (unsigned)(uintptr_t)(&WnT[0]);
  const char* gS = (const char*)wselfT;
  const char* gN = (const char*)wneighT;
#pragma unroll
  for (int i = 0; i < 8; ++i) {                             // 2048 b128 chunks / 256 thr
    unsigned off = (i * 256u + t) * 16u;
    async_ld_b128(ldsS + off, gS + off);
    async_ld_b128(ldsN + off, gN + off);
  }

  // cooperative fp32->bf16 staging of the 16x128 A tiles (h and agg), once per block
  {
    int row = (t * 8) >> 7;                                 // t*8 / 128
    int col = (t * 8) & 127;
    const float4* s0 = (const float4*)(hin + (size_t)(m0 + row) * D_GRAPH + col);
    float4 u = s0[0], v = s0[1];
    v8bf w;
    w[0] = (__bf16)u.x; w[1] = (__bf16)u.y; w[2] = (__bf16)u.z; w[3] = (__bf16)u.w;
    w[4] = (__bf16)v.x; w[5] = (__bf16)v.y; w[6] = (__bf16)v.z; w[7] = (__bf16)v.w;
    *(v8bf*)&Asf[row * D_GRAPH + col] = w;
    const float4* s1 = (const float4*)(agg + (size_t)(m0 + row) * D_GRAPH + col);
    float4 p = s1[0], q = s1[1];
    v8bf x;
    x[0] = (__bf16)p.x; x[1] = (__bf16)p.y; x[2] = (__bf16)p.z; x[3] = (__bf16)p.w;
    x[4] = (__bf16)q.x; x[5] = (__bf16)q.y; x[6] = (__bf16)q.z; x[7] = (__bf16)q.w;
    *(v8bf*)&Anb[row * D_GRAPH + col] = x;
  }
  wait_async0();
  __syncthreads();

  int lane = threadIdx.x & 31;
  int wv   = threadIdx.x >> 5;                              // 8 waves: one n-tile each
  int n0   = wv * 16;
  int mL   = lane & 15;
  int kha  = (lane >> 4) * 8;
  int nB   = n0 + (lane & 15);
  int khb  = (lane >> 4) * 8;

  v8f acc = {};
#pragma unroll
  for (int kt = 0; kt < D_GRAPH / 32; ++kt) {
    int k0 = kt * 32;
    v16bf as = frag_a_lds(Asf, mL, k0, kha);
    v16bf bs = frag_b_vec(WsT, D_GRAPH, nB, k0, khb);
    acc = __builtin_amdgcn_wmma_f32_16x16x32_bf16(false, as, false, bs, (short)0, acc, false, false);
    v16bf an = frag_a_lds(Anb, mL, k0, kha);
    v16bf bn = frag_b_vec(WnT, D_GRAPH, nB, k0, khb);
    acc = __builtin_amdgcn_wmma_f32_16x16x32_bf16(false, an, false, bn, (short)0, acc, false, false);
  }

  int n     = n0 + (lane & 15);
  int mbase = m0 + (lane >> 4) * 8;
  float bv  = bias[n];
#pragma unroll
  for (int r = 0; r < 8; ++r)
    hout[(size_t)(mbase + r) * D_GRAPH + n] = fmaxf(acc[r] + bv, 0.0f);
}

// ---------------- generic 1-wave WMMA GEMM tile -----------------------------
// C[m, n*cstride + coff] = post( pre(A) @ Bt^T + bias )
__global__ void __launch_bounds__(32)
k_gemm(const float* __restrict__ A, int lda, const __bf16* __restrict__ Bt, int ldk,
       const float* __restrict__ bias, float* __restrict__ C, int ldc,
       int cstride, int coff, int N, int K, int preact, int postact) {
  int lane = threadIdx.x;
  int m0 = blockIdx.x * 16, n0 = blockIdx.y * 16;
  int mA = m0 + (lane & 15);
  int kha = (lane >> 4) * 8;
  int nB  = n0 + (lane & 15);
  int khb = (lane >> 4) * 8;

  v8f acc = {};
  for (int k0 = 0; k0 < K; k0 += 32) {
    v16bf a = frag_a_f32(A, (size_t)lda, mA, k0, kha, preact);
    v16bf b = frag_b_vec(Bt, (size_t)ldk, nB, k0, khb);
    acc = __builtin_amdgcn_wmma_f32_16x16x32_bf16(false, a, false, b, (short)0, acc, false, false);
  }

  int n = n0 + (lane & 15);
  if (n < N) {
    int mbase = m0 + (lane >> 4) * 8;
    float bv = bias ? bias[n] : 0.0f;
#pragma unroll
    for (int r = 0; r < 8; ++r) {
      float v = acc[r] + bv;
      if (postact == 1) v = fmaxf(v, 0.0f);
      C[(size_t)(mbase + r) * ldc + (size_t)n * cstride + coff] = v;
    }
  }
}

__global__ void k_logsoftmax(const float* __restrict__ y, float* __restrict__ out) {
  int b = blockIdx.x * blockDim.x + threadIdx.x;
  if (b < NUM_G) {
    const float* row = y + (size_t)b * 16;
    float m = row[0];
#pragma unroll
    for (int j = 1; j < OUT_C; ++j) m = fmaxf(m, row[j]);
    float s = 0.0f;
#pragma unroll
    for (int j = 0; j < OUT_C; ++j) s += __expf(row[j] - m);
    float ls = m + __logf(s);
#pragma unroll
    for (int j = 0; j < OUT_C; ++j) out[(size_t)b * OUT_C + j] = row[j] - ls;
  }
}

// ---------------------------------------------------------------------------
extern "C" void kernel_launch(void* const* d_in, const int* in_sizes, int n_in,
                              void* d_out, int out_size, void* d_ws, size_t ws_size,
                              hipStream_t stream) {
  const int*   feat  = (const int*)  d_in[0];
  const int*   src   = (const int*)  d_in[1];
  const int*   dst   = (const int*)  d_in[2];
  const int*   gid   = (const int*)  d_in[3];
  const float* emb   = (const float*)d_in[4];
  const float* cWs   = (const float*)d_in[5];
  const float* cB    = (const float*)d_in[6];
  const float* cWn   = (const float*)d_in[7];
  const float* e1W   = (const float*)d_in[8];
  const float* e1b   = (const float*)d_in[9];
  const float* xw1   = (const float*)d_in[10];
  const float* xw2   = (const float*)d_in[11];
  const float* hW    = (const float*)d_in[12];
  const float* hb    = (const float*)d_in[13];
  const float* mW    = (const float*)d_in[14];
  const float* mb    = (const float*)d_in[15];
  const float* pW    = (const float*)d_in[16];
  const float* pb    = (const float*)d_in[17];
  float* out = (float*)d_out;

  size_t off = 0;
  auto alloc = [&](size_t bytes) -> void* {
    void* p = (char*)d_ws + off;
    off += (bytes + 255) & ~(size_t)255;
    return p;
  };
  float* hA     = (float*)alloc((size_t)N_NODES * D_GRAPH * 4);
  float* hB     = (float*)alloc((size_t)N_NODES * D_GRAPH * 4);
  float* agg    = (float*)alloc((size_t)N_NODES * D_GRAPH * 4);
  float* deg    = (float*)alloc((size_t)N_NODES * 4);
  float* dinv   = (float*)alloc((size_t)N_NODES * 4);
  float* hgsum  = (float*)alloc((size_t)NUM_G * D_GRAPH * 4);
  float* cnt    = (float*)alloc((size_t)NUM_G * 4);
  float* hg     = (float*)alloc((size_t)NUM_G * D_GRAPH * 4);
  float* hidden = (float*)alloc((size_t)NUM_G * HID * 4);
  float* h1     = (float*)alloc((size_t)NUM_G * INNER * 4);
  float* xcat   = (float*)alloc((size_t)NUM_G * HID * N_EXP * 4);
  float* t0     = (float*)alloc((size_t)NUM_G * HID * 4);
  float* t1     = (float*)alloc((size_t)NUM_G * HID * 4);
  float* ybuf   = (float*)alloc((size_t)NUM_G * 16 * 4);
  __bf16* wselfT = (__bf16*)alloc((size_t)3 * D_GRAPH * D_GRAPH * 2);
  __bf16* wneighT= (__bf16*)alloc((size_t)3 * D_GRAPH * D_GRAPH * 2);
  __bf16* e1wT   = (__bf16*)alloc((size_t)D_GRAPH * HID * 2);
  __bf16* xw1T   = (__bf16*)alloc((size_t)N_EXP * HID * INNER * 2);
  __bf16* xw2T   = (__bf16*)alloc((size_t)N_EXP * INNER * HID * 2);
  __bf16* hWT    = (__bf16*)alloc((size_t)HID * N_EXP * HID * 2);
  __bf16* mWT    = (__bf16*)alloc((size_t)L_MLP * HID * HID * 2);
  __bf16* pWT    = (__bf16*)alloc((size_t)16 * HID * 2);

  // --- degree / embedding ---------------------------------------------------
  hipMemsetAsync(deg, 0, (size_t)N_NODES * 4, stream);
  k_deg<<<(N_EDGES + 255) / 256, 256, 0, stream>>>(dst, deg, N_EDGES);
  k_deginv<<<(N_NODES + 255) / 256, 256, 0, stream>>>(deg, dinv, N_NODES);
  k_embed<<<(N_NODES * 32) / 256, 256, 0, stream>>>(feat, emb, hA);

  // --- weight conversion (fp32 -> transposed bf16) --------------------------
  auto conv_t = [&](const float* in, __bf16* o, int K, int N) {
    k_convert_t<<<(K * N + 255) / 256, 256, 0, stream>>>(in, o, K, N);
  };
  for (int l = 0; l < 3; ++l) {
    conv_t(cWs + (size_t)l * D_GRAPH * D_GRAPH, wselfT + (size_t)l * D_GRAPH * D_GRAPH, D_GRAPH, D_GRAPH);
    conv_t(cWn + (size_t)l * D_GRAPH * D_GRAPH, wneighT + (size_t)l * D_GRAPH * D_GRAPH, D_GRAPH, D_GRAPH);
  }
  conv_t(e1W, e1wT, D_GRAPH, HID);
  for (int e = 0; e < N_EXP; ++e) {
    conv_t(xw1 + (size_t)e * HID * INNER, xw1T + (size_t)e * HID * INNER, HID, INNER);
    conv_t(xw2 + (size_t)e * INNER * HID, xw2T + (size_t)e * INNER * HID, INNER, HID);
  }
  conv_t(hW, hWT, HID * N_EXP, HID);
  for (int l = 0; l < L_MLP; ++l)
    conv_t(mW + (size_t)l * HID * HID, mWT + (size_t)l * HID * HID, HID, HID);
  k_convert_pred<<<(16 * HID + 255) / 256, 256, 0, stream>>>(pW, pWT);

  // --- 3 GraphSAGE conv layers (ping-pong hA <-> hB) ------------------------
  float* hin = hA; float* hout = hB;
  for (int l = 0; l < 3; ++l) {
    hipMemsetAsync(agg, 0, (size_t)N_NODES * D_GRAPH * 4, stream);
    k_edge_scatter<<<2048, 256, 0, stream>>>(hin, src, dst, agg, N_EDGES);
    k_scale_agg<<<(N_NODES * D_GRAPH) / 256, 256, 0, stream>>>(agg, dinv);
    k_conv<<<N_NODES / 16, 256, 0, stream>>>(
        hin, agg,
        wselfT + (size_t)l * D_GRAPH * D_GRAPH,
        wneighT + (size_t)l * D_GRAPH * D_GRAPH,
        cB + (size_t)l * D_GRAPH, hout);
    float* tmp = hin; hin = hout; hout = tmp;
  }
  float* hfin = hin;   // after 3 swaps the result is in `hin`

  // --- per-graph mean pooling ----------------------------------------------
  hipMemsetAsync(hgsum, 0, (size_t)NUM_G * D_GRAPH * 4, stream);
  hipMemsetAsync(cnt, 0, (size_t)NUM_G * 4, stream);
  k_pool_accum<<<(N_NODES * 32) / 256, 256, 0, stream>>>(hfin, gid, hgsum, cnt);
  k_pool_final<<<(NUM_G * D_GRAPH) / 256, 256, 0, stream>>>(hgsum, cnt, hg);

  // --- dense head -----------------------------------------------------------
  // hidden = relu(hg @ e1W + e1b)
  k_gemm<<<dim3(NUM_G / 16, HID / 16), 32, 0, stream>>>(
      hg, D_GRAPH, e1wT, D_GRAPH, e1b, hidden, HID, 1, 0, HID, D_GRAPH, 0, 1);
  hipMemcpyAsync(out + NUM_G * OUT_C, hidden, (size_t)NUM_G * HID * 4,
                 hipMemcpyDeviceToDevice, stream);   // second output: hidden

  // MoE experts, interleaved concat x[b, h*5 + e]
  for (int e = 0; e < N_EXP; ++e) {
    k_gemm<<<dim3(NUM_G / 16, INNER / 16), 32, 0, stream>>>(
        hidden, HID, xw1T + (size_t)e * HID * INNER, HID, nullptr,
        h1, INNER, 1, 0, INNER, HID, 0, 1);
    k_gemm<<<dim3(NUM_G / 16, HID / 16), 32, 0, stream>>>(
        h1, INNER, xw2T + (size_t)e * INNER * HID, INNER, nullptr,
        xcat, HID * N_EXP, N_EXP, e, HID, INNER, 0, 0);
  }
  // head
  k_gemm<<<dim3(NUM_G / 16, HID / 16), 32, 0, stream>>>(
      xcat, HID * N_EXP, hWT, HID * N_EXP, hb, t0, HID, 1, 0, HID, HID * N_EXP, 0, 0);
  // MLP: x = leaky(x) @ W + b
  float* tin = t0; float* tout = t1;
  for (int l = 0; l < L_MLP; ++l) {
    k_gemm<<<dim3(NUM_G / 16, HID / 16), 32, 0, stream>>>(
        tin, HID, mWT + (size_t)l * HID * HID, HID, mb + (size_t)l * HID,
        tout, HID, 1, 0, HID, HID, 1, 0);
    float* tmp = tin; tin = tout; tout = tmp;
  }
  // pred: y = leaky(x) @ pred_W + pred_b  (N=10 padded to 16)
  k_gemm<<<dim3(NUM_G / 16, 1), 32, 0, stream>>>(
      tin, HID, pWT, HID, pb, ybuf, 16, 1, 0, OUT_C, HID, 1, 0);
  k_logsoftmax<<<(NUM_G + 255) / 256, 256, 0, stream>>>(ybuf, out);
}
